// Deepseekv3MoE_70016556860062
// MI455X (gfx1250) — matrix operand order repaired and checked by reading, not compiled, not run on hardware
//
#include <hip/hip_runtime.h>
#include <hip/hip_bf16.h>
#include <cstdint>

// ---------------- problem constants ----------------
#define EXPERTS 32
#define TOPK    4
#define HID     1024
#define INTER   768
#define TOK     4096
#define NGROUP  8
#define TOPKG   4
#define CAP     1024

#define KC    128            // K-chunk staged in LDS per barrier interval
#define BSTR  144            // padded LDS row stride in elements (288 B)

typedef __bf16 bf16;
typedef __attribute__((ext_vector_type(8)))  bf16  v8bf;
typedef __attribute__((ext_vector_type(16))) bf16  v16bf;
typedef __attribute__((ext_vector_type(8)))  float v8f;

// ---------------- workspace layout (bytes) ----------------
static constexpr size_t O_COUNTS = 0;                                     // 32 ints
static constexpr size_t O_SCORES = 256;                                   // T*E f32
static constexpr size_t O_SEL    = O_SCORES + (size_t)TOK*EXPERTS*4;      // T*K int
static constexpr size_t O_RW     = O_SEL    + (size_t)TOK*TOPK*4;         // T*K f32
static constexpr size_t O_SLOT   = O_RW     + (size_t)TOK*TOPK*4;         // T*K int
static constexpr size_t O_BUF    = (O_SLOT + (size_t)TOK*TOPK*4 + 255) & ~(size_t)255; // E*CAP*H bf16
static constexpr size_t O_ACT    = O_BUF + (size_t)EXPERTS*CAP*HID*2;     // E*CAP*I bf16
static constexpr size_t O_Y      = O_ACT + (size_t)EXPERTS*CAP*INTER*2;   // E*CAP*H f32
static constexpr size_t O_W1B    = O_Y   + (size_t)EXPERTS*CAP*HID*4;     // E*I*H bf16
static constexpr size_t O_W3B    = O_W1B + (size_t)EXPERTS*INTER*HID*2;
static constexpr size_t O_W2B    = O_W3B + (size_t)EXPERTS*INTER*HID*2;   // E*H*I bf16

// ---------------- helpers ----------------
__device__ __forceinline__ float dev_sigmoid(float x) {
    return 1.0f / (1.0f + __expf(-x));
}

// A fragment, 16-bit 16x32 layout: lane L holds row (L&15); K in
// [(L>>4)*16, +16) contiguous for elements 0..15.
__device__ __forceinline__ v16bf load_a_frag(const bf16* __restrict__ p, int kb) {
    v8bf lo = *(const v8bf*)(p + kb);
    v8bf hi = *(const v8bf*)(p + kb + 8);
    v16bf r;
#pragma unroll
    for (int i = 0; i < 8; ++i) { r[i] = lo[i]; r[i + 8] = hi[i]; }
    return r;
}

#define WMMA_BF16(a, b, c) \
    __builtin_amdgcn_wmma_f32_16x16x32_bf16(false, (a), false, (b), (short)0, (c), false, false)

// ---------------- kernels ----------------
__global__ void k_zero_counts(int* __restrict__ counts) {
    if (threadIdx.x < EXPERTS) counts[threadIdx.x] = 0;
}

// scores[t][e] = sigmoid( x[t] . gate_w[e] ), one block per token
__global__ void k_router_scores(const float* __restrict__ x,
                                const float* __restrict__ gw,
                                float* __restrict__ scores) {
    __shared__ float xs[HID];
    __shared__ float partial[8][EXPERTS];
    const int t = blockIdx.x;
    for (int i = threadIdx.x; i < HID; i += 256) xs[i] = x[(size_t)t * HID + i];
    __syncthreads();
    const int e = threadIdx.x & 31;
    const int part = threadIdx.x >> 5;          // 0..7
    const float* g = gw + (size_t)e * HID + part * 128;
    float s = 0.0f;
#pragma unroll 8
    for (int i = 0; i < 128; ++i) s += xs[part * 128 + i] * g[i];
    partial[part][e] = s;
    __syncthreads();
    if (part == 0) {
        float tot = 0.0f;
#pragma unroll
        for (int p = 0; p < 8; ++p) tot += partial[p][e];
        scores[(size_t)t * EXPERTS + e] = dev_sigmoid(tot);
    }
}

// group-limited top-k routing, one thread per token
__global__ void k_router_topk(const float* __restrict__ scores,
                              const float* __restrict__ bias,
                              int* __restrict__ sel, float* __restrict__ rw,
                              int* __restrict__ counts, int* __restrict__ slotArr) {
    const int t = blockIdx.x * blockDim.x + threadIdx.x;
    if (t >= TOK) return;
    float sc[EXPERTS], sfc[EXPERTS];
    for (int e = 0; e < EXPERTS; ++e) {
        sc[e]  = scores[(size_t)t * EXPERTS + e];
        sfc[e] = sc[e] + bias[e];
    }
    // per-group sum of top-2 (groups of E/NGROUP = 4)
    float gs[NGROUP];
    for (int g = 0; g < NGROUP; ++g) {
        float m1 = -1e30f, m2 = -1e30f;
        for (int j = 0; j < EXPERTS / NGROUP; ++j) {
            float v = sfc[g * (EXPERTS / NGROUP) + j];
            if (v > m1) { m2 = m1; m1 = v; }
            else if (v > m2) { m2 = v; }
        }
        gs[g] = m1 + m2;
    }
    // top-4 groups
    bool gmask[NGROUP];
    for (int g = 0; g < NGROUP; ++g) gmask[g] = false;
    for (int it = 0; it < TOPKG; ++it) {
        int   bi = -1; float bv = -1e30f;
        for (int g = 0; g < NGROUP; ++g)
            if (!gmask[g] && gs[g] > bv) { bv = gs[g]; bi = g; }
        gmask[bi] = true;
    }
    // top-4 experts among masked (sfc = sigmoid+bias > 0 always)
    float tmp[EXPERTS];
    for (int e = 0; e < EXPERTS; ++e)
        tmp[e] = gmask[e / (EXPERTS / NGROUP)] ? sfc[e] : 0.0f;
    int   se[TOPK];
    float wv[TOPK];
    float wsum = 0.0f;
    for (int k = 0; k < TOPK; ++k) {
        int bi = 0; float bv = -1e30f;
        for (int e = 0; e < EXPERTS; ++e)
            if (tmp[e] > bv) { bv = tmp[e]; bi = e; }
        se[k] = bi;
        wv[k] = sc[bi];           // raw sigmoid score (no bias)
        wsum += wv[k];
        tmp[bi] = -1e30f;
    }
    const float inv = 1.0f / wsum;
    for (int k = 0; k < TOPK; ++k) {
        const int n = t * TOPK + k;
        sel[n] = se[k];
        rw[n]  = wv[k] * inv;
        const int p = atomicAdd(&counts[se[k]], 1);
        slotArr[n] = p;
    }
}

// copy token row -> per-expert bf16 buffer, one block per (token,k)
__global__ void k_gather(const float* __restrict__ x,
                         const int* __restrict__ sel,
                         const int* __restrict__ slotArr,
                         bf16* __restrict__ buf) {
    const int n = blockIdx.x;
    const int p = slotArr[n];
    if (p >= CAP) return;
    const int e = sel[n];
    const int t = n >> 2;
    const float* xr = x + (size_t)t * HID;
    bf16* br = buf + ((size_t)e * CAP + p) * HID;
    for (int i = threadIdx.x; i < HID; i += blockDim.x) br[i] = (bf16)xr[i];
}

__global__ void k_cvt_bf16(const float* __restrict__ src, bf16* __restrict__ dst, size_t n) {
    size_t i = blockIdx.x * (size_t)blockDim.x + threadIdx.x;
    const size_t stride = gridDim.x * (size_t)blockDim.x;
    for (; i < n; i += stride) dst[i] = (bf16)src[i];
}

// act[e,c,i] = silu(buf@w1^T) * (buf@w3^T)
// 4-wave block, 128x32 macro-tile. B-tiles (w1,w3: 32 rows x KC) staged in
// LDS once per K-chunk and shared by all 4 waves (4x less weight traffic);
// A streams from global. 8 WMMAs per K-step per wave.
__global__ void __launch_bounds__(128)
k_gemm1(const bf16* __restrict__ buf, const bf16* __restrict__ w1b,
        const bf16* __restrict__ w3b, bf16* __restrict__ act,
        const int* __restrict__ counts) {
    __shared__ __align__(16) bf16 lB1[32 * BSTR];
    __shared__ __align__(16) bf16 lB3[32 * BSTR];
    const int it = blockIdx.x;                 // i macro-tile (32 wide)
    const int rt = blockIdx.y;                 // row macro-tile (128 rows)
    const int e  = blockIdx.z;
    if (rt * 128 >= counts[e]) return;         // block-uniform
    const int tid  = threadIdx.x;
    const int wave = tid >> 5;
    const int lane = tid & 31;
    const int half = lane >> 4;
    const int l15  = lane & 15;
    const bf16* a0 = buf + ((size_t)e * CAP + rt * 128 + wave * 32 + l15) * HID;
    const bf16* a1 = a0 + (size_t)16 * HID;
    const bf16* w1base = w1b + ((size_t)e * INTER + it * 32) * HID;
    const bf16* w3base = w3b + ((size_t)e * INTER + it * 32) * HID;
    const int kbA = half * 16;   // A: contiguous 16 elements
    const int kbB = half * 16;   // B: contiguous 16 elements
    v8f acc1[2][2] = {};
    v8f acc3[2][2] = {};
    for (int k0 = 0; k0 < HID; k0 += KC) {
        __syncthreads();                       // prior chunk fully consumed
        // stage 32 rows x KC of w1 and w3 (16B chunks, 512 each)
#pragma unroll
        for (int i = 0; i < 4; ++i) {
            const int idx = tid + i * 128;     // 0..511
            const int row = idx >> 4;          // KC/8 = 16 chunks per row
            const int co  = (idx & 15) * 8;
            *(v8bf*)&lB1[row * BSTR + co] = *(const v8bf*)(w1base + (size_t)row * HID + k0 + co);
            *(v8bf*)&lB3[row * BSTR + co] = *(const v8bf*)(w3base + (size_t)row * HID + k0 + co);
        }
        __syncthreads();
#pragma unroll
        for (int kk = 0; kk < KC; kk += 32) {
            const v16bf A0  = load_a_frag(a0 + k0 + kk, kbA);
            const v16bf A1  = load_a_frag(a1 + k0 + kk, kbA);
            const v16bf B10 = *(const v16bf*)&lB1[l15 * BSTR + kk + kbB];
            const v16bf B11 = *(const v16bf*)&lB1[(16 + l15) * BSTR + kk + kbB];
            const v16bf B30 = *(const v16bf*)&lB3[l15 * BSTR + kk + kbB];
            const v16bf B31 = *(const v16bf*)&lB3[(16 + l15) * BSTR + kk + kbB];
            acc1[0][0] = WMMA_BF16(A0, B10, acc1[0][0]);
            acc1[0][1] = WMMA_BF16(A0, B11, acc1[0][1]);
            acc1[1][0] = WMMA_BF16(A1, B10, acc1[1][0]);
            acc1[1][1] = WMMA_BF16(A1, B11, acc1[1][1]);
            acc3[0][0] = WMMA_BF16(A0, B30, acc3[0][0]);
            acc3[0][1] = WMMA_BF16(A0, B31, acc3[0][1]);
            acc3[1][0] = WMMA_BF16(A1, B30, acc3[1][0]);
            acc3[1][1] = WMMA_BF16(A1, B31, acc3[1][1]);
        }
    }
    // D layout: VGPR r = row (r + 8*half), col = l15
#pragma unroll
    for (int mi = 0; mi < 2; ++mi) {
#pragma unroll
        for (int ni = 0; ni < 2; ++ni) {
#pragma unroll
            for (int r = 0; r < 8; ++r) {
                const float h1 = acc1[mi][ni][r];
                const float h3 = acc3[mi][ni][r];
                const float v  = (h1 * dev_sigmoid(h1)) * h3;
                const int row = rt * 128 + wave * 32 + mi * 16 + r + half * 8;
                const int col = it * 32 + ni * 16 + l15;
                act[((size_t)e * CAP + row) * INTER + col] = (bf16)v;
            }
        }
    }
}

// y[e,c,h] = act @ w2^T — 4-wave block, 128x64 macro-tile, LDS-shared B
__global__ void __launch_bounds__(128)
k_gemm2(const bf16* __restrict__ act, const bf16* __restrict__ w2b,
        float* __restrict__ y, const int* __restrict__ counts) {
    __shared__ __align__(16) bf16 lB[64 * BSTR];
    const int ht = blockIdx.x;                 // h macro-tile (64 wide)
    const int rt = blockIdx.y;                 // row macro-tile (128 rows)
    const int e  = blockIdx.z;
    if (rt * 128 >= counts[e]) return;         // block-uniform
    const int tid  = threadIdx.x;
    const int wave = tid >> 5;
    const int lane = tid & 31;
    const int half = lane >> 4;
    const int l15  = lane & 15;
    const bf16* a0 = act + ((size_t)e * CAP + rt * 128 + wave * 32 + l15) * INTER;
    const bf16* a1 = a0 + (size_t)16 * INTER;
    const bf16* wbase = w2b + ((size_t)e * HID + ht * 64) * INTER;
    const int kbA = half * 16;
    const int kbB = half * 16;
    v8f acc[2][4] = {};
    for (int k0 = 0; k0 < INTER; k0 += KC) {
        __syncthreads();
        // stage 64 rows x KC of w2 (16B chunks, 1024)
#pragma unroll
        for (int i = 0; i < 8; ++i) {
            const int idx = tid + i * 128;     // 0..1023
            const int row = idx >> 4;
            const int co  = (idx & 15) * 8;
            *(v8bf*)&lB[row * BSTR + co] = *(const v8bf*)(wbase + (size_t)row * INTER + k0 + co);
        }
        __syncthreads();
#pragma unroll
        for (int kk = 0; kk < KC; kk += 32) {
            const v16bf A0 = load_a_frag(a0 + k0 + kk, kbA);
            const v16bf A1 = load_a_frag(a1 + k0 + kk, kbA);
            v16bf B[4];
#pragma unroll
            for (int ni = 0; ni < 4; ++ni)
                B[ni] = *(const v16bf*)&lB[(ni * 16 + l15) * BSTR + kk + kbB];
#pragma unroll
            for (int ni = 0; ni < 4; ++ni) {
                acc[0][ni] = WMMA_BF16(A0, B[ni], acc[0][ni]);
                acc[1][ni] = WMMA_BF16(A1, B[ni], acc[1][ni]);
            }
        }
    }
#pragma unroll
    for (int mi = 0; mi < 2; ++mi) {
#pragma unroll
        for (int ni = 0; ni < 4; ++ni) {
#pragma unroll
            for (int r = 0; r < 8; ++r) {
                const int row = rt * 128 + wave * 32 + mi * 16 + r + half * 8;
                const int col = ht * 64 + ni * 16 + l15;
                y[((size_t)e * CAP + row) * HID + col] = acc[mi][ni][r];
            }
        }
    }
}

// out[t] = sum_k rw[t,k] * y[sel, slot] — one block per token
__global__ void k_combine(const float* __restrict__ y,
                          const int* __restrict__ sel,
                          const int* __restrict__ slotArr,
                          const float* __restrict__ rw,
                          float* __restrict__ out) {
    const int t = blockIdx.x;
    const int tid = threadIdx.x;               // 256
    float acc[HID / 256];
#pragma unroll
    for (int j = 0; j < HID / 256; ++j) acc[j] = 0.0f;
    for (int k = 0; k < TOPK; ++k) {
        const int n = t * TOPK + k;
        const int p = slotArr[n];
        if (p >= CAP) continue;
        const int e = sel[n];
        const float w = rw[n];
        const float* yr = y + ((size_t)e * CAP + p) * HID;
#pragma unroll
        for (int j = 0; j < HID / 256; ++j) acc[j] += w * yr[tid + j * 256];
    }
#pragma unroll
    for (int j = 0; j < HID / 256; ++j) out[(size_t)t * HID + tid + j * 256] = acc[j];
}

// ---------------- launch ----------------
extern "C" void kernel_launch(void* const* d_in, const int* in_sizes, int n_in,
                              void* d_out, int out_size, void* d_ws, size_t ws_size,
                              hipStream_t stream) {
    const float* x    = (const float*)d_in[0];   // [T,H]
    const float* gw   = (const float*)d_in[1];   // [E,H]
    const float* w1   = (const float*)d_in[2];   // [E,I,H]
    const float* w3   = (const float*)d_in[3];   // [E,I,H]
    const float* w2   = (const float*)d_in[4];   // [E,H,I]
    const float* bias = (const float*)d_in[5];   // [E]
    float* out = (float*)d_out;

    char* ws = (char*)d_ws;
    int*   counts  = (int*)  (ws + O_COUNTS);
    float* scores  = (float*)(ws + O_SCORES);
    int*   sel     = (int*)  (ws + O_SEL);
    float* rw      = (float*)(ws + O_RW);
    int*   slotArr = (int*)  (ws + O_SLOT);
    bf16*  buf     = (bf16*) (ws + O_BUF);
    bf16*  act     = (bf16*) (ws + O_ACT);
    float* y       = (float*)(ws + O_Y);
    bf16*  w1b     = (bf16*) (ws + O_W1B);
    bf16*  w3b     = (bf16*) (ws + O_W3B);
    bf16*  w2b     = (bf16*) (ws + O_W2B);

    k_zero_counts<<<1, 32, 0, stream>>>(counts);

    // weight conversion fp32 -> bf16 (151 MB total: fits global L2)
    const size_t nW = (size_t)EXPERTS * INTER * HID;
    k_cvt_bf16<<<4096, 256, 0, stream>>>(w1, w1b, nW);
    k_cvt_bf16<<<4096, 256, 0, stream>>>(w3, w3b, nW);
    k_cvt_bf16<<<4096, 256, 0, stream>>>(w2, w2b, nW);

    k_router_scores<<<TOK, 256, 0, stream>>>(x, gw, scores);
    k_router_topk<<<TOK / 256, 256, 0, stream>>>(scores, bias, sel, rw, counts, slotArr);
    k_gather<<<TOK * TOPK, 256, 0, stream>>>(x, sel, slotArr, buf);

    dim3 g1(INTER / 32, CAP / 128, EXPERTS);
    k_gemm1<<<g1, 128, 0, stream>>>(buf, w1b, w3b, act, counts);

    dim3 g2(HID / 64, CAP / 128, EXPERTS);
    k_gemm2<<<g2, 128, 0, stream>>>(act, w2b, y, counts);

    k_combine<<<TOK, 256, 0, stream>>>(y, sel, slotArr, rw, out);
}
